// MultiQueryAttention_35046933135862
// MI455X (gfx1250) — matrix operand order repaired and checked by reading, not compiled, not run on hardware
//
#include <hip/hip_runtime.h>
#include <hip/hip_bf16.h>

// ---------------------------------------------------------------------------
// Multi-query attention for MI455X (gfx1250, wave32, WMMA 16x16x32 bf16).
//   q  = (x @ Wq) * hd^-0.5        -> bf16, layout (b,h,n,hd)
//   kv = x @ Wkv; K (b,t,hd) row-major bf16, V stored transposed (b,hd,t)
//   y  = flash-attention(q,K,V)    -> bf16, layout (b,n,d)
//   out= y @ Wo                    -> f32
// Matrix math via v_wmma_f32_16x16x32_bf16; flash kernel streams K/V tiles
// into LDS with GLOBAL_LOAD_ASYNC_TO_LDS_B128 (double-buffered, ASYNCcnt).
// ---------------------------------------------------------------------------

typedef __attribute__((ext_vector_type(16))) __bf16 v16bf;
typedef __attribute__((ext_vector_type(8)))  __bf16 v8bf;
typedef __attribute__((ext_vector_type(8)))  float  v8f;

#define BATCH 4
#define SEQ   2048
#define DMODEL 2048
#define NHEAD 16
#define HDIM  128

// ---------------------------------------------------------------- converts
__global__ __launch_bounds__(256) void cvt_bf16_kernel(const float* __restrict__ in,
                                                       __bf16* __restrict__ out, long n) {
  long i = (long)blockIdx.x * blockDim.x + threadIdx.x;
  long stride = (long)gridDim.x * blockDim.x;
  for (; i < n; i += stride) out[i] = (__bf16)in[i];
}

// out[c*rows + r] = in[r*cols + c]  (K-major "B-transposed" operand for WMMA)
__global__ __launch_bounds__(256) void cvt_transpose_kernel(const float* __restrict__ in,
                                                            __bf16* __restrict__ out,
                                                            int rows, int cols) {
  long total = (long)rows * cols;
  long i = (long)blockIdx.x * blockDim.x + threadIdx.x;
  long stride = (long)gridDim.x * blockDim.x;
  for (; i < total; i += stride) {
    int r = (int)(i / cols);
    int c = (int)(i % cols);
    out[(size_t)c * rows + r] = (__bf16)in[i];
  }
}

// ---------------------------------------------------------------- WMMA GEMM
// C[M,N] = A[M,K] @ B[K,N], A row-major bf16, BT = B^T row-major bf16.
// 256 threads = 8 waves (4 M x 2 N); block tile 64x128, wave tile 16x64.
// MODE 0: f32 store (row-major)                        [y @ Wo -> d_out]
// MODE 1: scale + bf16 store, (m,j)->(b,h,n,hd)        [x @ Wq]
// MODE 2: bf16 split: j<128 -> K (b,t,hd); else V^T (b,hd,t)
template <int MODE>
__global__ __launch_bounds__(256) void gemm_wmma_kernel(
    const __bf16* __restrict__ A, const __bf16* __restrict__ BT,
    float* __restrict__ Cf, __bf16* __restrict__ O1, __bf16* __restrict__ O2,
    int M, int N, int K, float scale)
{
  const int lane   = threadIdx.x & 31;
  const int lane16 = lane & 15;
  const int laneHi = lane >> 4;
  const int w  = threadIdx.x >> 5;
  const int wm = w >> 1;
  const int wn = w & 1;
  const int m0 = blockIdx.y * 64 + wm * 16;
  const int n0 = blockIdx.x * 128 + wn * 64;

  v8f acc[4];
#pragma unroll
  for (int f = 0; f < 4; ++f) acc[f] = (v8f)0.0f;

  const __bf16* aRow = A + (size_t)(m0 + lane16) * K;

  for (int k0 = 0; k0 < K; k0 += 32) {
    const __bf16* pa = aRow + k0 + laneHi * 16;
    v8bf alo = *(const v8bf*)pa;
    v8bf ahi = *(const v8bf*)(pa + 8);
    v16bf aF;
#pragma unroll
    for (int i = 0; i < 8; ++i) { aF[i] = alo[i]; aF[8 + i] = ahi[i]; }

    if (k0 + 32 < K) __builtin_prefetch(pa + 32, 0, 1);

#pragma unroll
    for (int f = 0; f < 4; ++f) {
      const int col = n0 + f * 16 + lane16;
      v16bf bF = *(const v16bf*)(BT + (size_t)col * K + k0 + laneHi * 16);
      acc[f] = __builtin_amdgcn_wmma_f32_16x16x32_bf16(
          false, aF, false, bF, (short)0, acc[f], false, false);
    }
  }

#pragma unroll
  for (int f = 0; f < 4; ++f) {
#pragma unroll
    for (int r = 0; r < 8; ++r) {
      const int m = m0 + r + 8 * laneHi;
      const int j = n0 + f * 16 + lane16;
      const float v = acc[f][r];
      if (MODE == 0) {
        Cf[(size_t)m * N + j] = v;
      } else if (MODE == 1) {
        const int b = m >> 11, n = m & 2047;
        const int h = j >> 7,  hd = j & 127;
        O1[(((size_t)b * NHEAD + h) * SEQ + n) * HDIM + hd] = (__bf16)(v * scale);
      } else {
        const int b = m >> 11, t = m & 2047;
        if (j < HDIM) O1[((size_t)b * SEQ + t) * HDIM + j] = (__bf16)v;
        else          O2[((size_t)b * HDIM + (j - HDIM)) * SEQ + t] = (__bf16)v;
      }
    }
  }
}

// ------------------------------------------------------- async copy helpers
// Per-lane async DMA of 16 bytes global -> LDS (tracked by ASYNCcnt).
// LDS byte address = low 32 bits of the generic pointer.
__device__ __forceinline__ void async_copy_b128(const __bf16* gptr, __bf16* lptr) {
  unsigned lds_addr = (unsigned)(unsigned long long)lptr;
  asm volatile("global_load_async_to_lds_b128 %0, %1, off"
               :: "v"(lds_addr), "v"(gptr) : "memory");
}
__device__ __forceinline__ void wait_async0() {
  asm volatile("s_wait_asynccnt 0x0" ::: "memory");
}

// ---------------------------------------------------------------- flash MQA
// 1 block = 4 waves sharing one (b,h); each wave owns 16 query rows.
// K/V tiles (32 columns) are staged into LDS once per block with async
// loads, double-buffered so the copy of step i+1 overlaps the 16 WMMAs of
// step i. P goes through LDS to convert C-layout f32 -> A-layout bf16.
__global__ __launch_bounds__(128) void mqa_flash_kernel(
    const __bf16* __restrict__ Q,    // (b,h,n,hd), pre-scaled
    const __bf16* __restrict__ Kb,   // (b,t,hd) row-major
    const __bf16* __restrict__ VT,   // (b,hd,t)
    __bf16* __restrict__ Y)          // (b,n,d)
{
  __shared__ __align__(64) __bf16 ldsK[2][32][HDIM];  // 2 x 8 KB
  __shared__ __align__(64) __bf16 ldsV[2][HDIM][32];  // 2 x 8 KB
  __shared__ __align__(64) __bf16 ldsP[4][16][32];    // 4 KB

  const int tid    = threadIdx.x;
  const int lane   = tid & 31;
  const int lane16 = lane & 15;
  const int laneHi = lane >> 4;
  const int warp   = tid >> 5;

  const int blk = blockIdx.x;          // 0..2047; 32 blocks per (b,h)
  const int b   = blk >> 9;
  const int h   = (blk >> 5) & 15;
  const int m0  = ((blk & 31) * 4 + warp) * 16;

  const __bf16* qBase = Q  + (((size_t)b * NHEAD + h) * SEQ) * HDIM;
  const __bf16* kBase = Kb + (size_t)b * SEQ * HDIM;
  const __bf16* vBase = VT + (size_t)b * HDIM * SEQ;

  // stage one 32-column K/V tile into LDS buffer `buf` (8 B128 per thread)
  auto stage = [&](int buf, int t0) {
    const __bf16* gK = kBase + (size_t)t0 * HDIM;   // 8 KB contiguous
#pragma unroll
    for (int i = 0; i < 4; ++i) {
      int chunk = tid + i * 128;                    // 0..511, 16B each
      async_copy_b128(gK + chunk * 8, &ldsK[buf][0][0] + chunk * 8);
    }
#pragma unroll
    for (int i = 0; i < 4; ++i) {
      int chunk = tid + i * 128;
      int row = chunk >> 2, seg = chunk & 3;        // 128 rows x 4 segs
      async_copy_b128(vBase + (size_t)row * SEQ + t0 + seg * 8,
                      &ldsV[buf][row][seg * 8]);
    }
  };

  // Q A-fragments: 16 rows x 128 K = 4 chunks of K=32
  v16bf qF[4];
#pragma unroll
  for (int c = 0; c < 4; ++c) {
    const __bf16* p = qBase + (size_t)(m0 + lane16) * HDIM + c * 32 + laneHi * 16;
    v8bf lo = *(const v8bf*)p;
    v8bf hi = *(const v8bf*)(p + 8);
#pragma unroll
    for (int i = 0; i < 8; ++i) { qF[c][i] = lo[i]; qF[c][8 + i] = hi[i]; }
  }

  float mi[8], li[8];
  v8f acc[8];
#pragma unroll
  for (int r = 0; r < 8; ++r) { mi[r] = -3.0e30f; li[r] = 0.0f; }
#pragma unroll
  for (int f = 0; f < 8; ++f) acc[f] = (v8f)0.0f;

  stage(0, 0);
  wait_async0();
  __syncthreads();

  for (int t0 = 0; t0 < SEQ; t0 += 32) {
    const int cur = (t0 >> 5) & 1;
    if (t0 + 32 < SEQ) stage(cur ^ 1, t0 + 32);     // overlap next copy

    // ---- scores: S = q @ K^T for 2 column tiles of 16 (K frags from LDS)
    v8f s0 = (v8f)0.0f, s1 = (v8f)0.0f;
#pragma unroll
    for (int c = 0; c < 4; ++c) {
      v16bf k0F = *(const v16bf*)&ldsK[cur][lane16][c * 32 + laneHi * 16];
      v16bf k1F = *(const v16bf*)&ldsK[cur][16 + lane16][c * 32 + laneHi * 16];
      s0 = __builtin_amdgcn_wmma_f32_16x16x32_bf16(false, qF[c], false, k0F, (short)0, s0, false, false);
      s1 = __builtin_amdgcn_wmma_f32_16x16x32_bf16(false, qF[c], false, k1F, (short)0, s1, false, false);
    }

    // ---- online softmax (row = r + 8*laneHi; reduce over 16-lane half)
    float mnew[8], alpha[8];
#pragma unroll
    for (int r = 0; r < 8; ++r) {
      float mx = fmaxf(s0[r], s1[r]);
#pragma unroll
      for (int d = 8; d >= 1; d >>= 1) mx = fmaxf(mx, __shfl_xor(mx, d, 32));
      mnew[r]  = fmaxf(mi[r], mx);
      alpha[r] = __expf(mi[r] - mnew[r]);
      li[r]   *= alpha[r];
    }
#pragma unroll
    for (int f = 0; f < 8; ++f)
#pragma unroll
      for (int r = 0; r < 8; ++r) acc[f][r] *= alpha[r];

    float p0[8], p1[8];
#pragma unroll
    for (int r = 0; r < 8; ++r) {
      p0[r] = __expf(s0[r] - mnew[r]);
      p1[r] = __expf(s1[r] - mnew[r]);
      float sum = p0[r] + p1[r];
#pragma unroll
      for (int d = 8; d >= 1; d >>= 1) sum += __shfl_xor(sum, d, 32);
      li[r] += sum;
      mi[r]  = mnew[r];
    }

    // ---- C-layout -> A-layout via LDS (same-wave; compiler inserts dscnt waits)
#pragma unroll
    for (int r = 0; r < 8; ++r) {
      ldsP[warp][r + 8 * laneHi][lane16]      = (__bf16)p0[r];
      ldsP[warp][r + 8 * laneHi][16 + lane16] = (__bf16)p1[r];
    }
    const __bf16* pp = &ldsP[warp][lane16][laneHi * 16];
    v8bf plo = *(const v8bf*)pp;
    v8bf phi = *(const v8bf*)(pp + 8);
    v16bf pF;
#pragma unroll
    for (int i = 0; i < 8; ++i) { pF[i] = plo[i]; pF[8 + i] = phi[i]; }

    // ---- y += P @ V : 8 hd tiles (V frags from LDS, contiguous 32B)
#pragma unroll
    for (int f = 0; f < 8; ++f) {
      v16bf vF = *(const v16bf*)&ldsV[cur][f * 16 + lane16][laneHi * 16];
      acc[f] = __builtin_amdgcn_wmma_f32_16x16x32_bf16(false, pF, false, vF, (short)0, acc[f], false, false);
    }

    wait_async0();        // this wave's next-tile copies have landed
    __syncthreads();      // everyone done reading `cur`, next tile visible
  }

  float inv[8];
#pragma unroll
  for (int r = 0; r < 8; ++r) inv[r] = 1.0f / li[r];
#pragma unroll
  for (int f = 0; f < 8; ++f)
#pragma unroll
    for (int r = 0; r < 8; ++r) {
      const int n = m0 + r + 8 * laneHi;
      Y[((size_t)b * SEQ + n) * DMODEL + h * HDIM + f * 16 + lane16] =
          (__bf16)(acc[f][r] * inv[r]);
    }
}

// ---------------------------------------------------------------- launch
extern "C" void kernel_launch(void* const* d_in, const int* in_sizes, int n_in,
                              void* d_out, int out_size, void* d_ws, size_t ws_size,
                              hipStream_t stream) {
  (void)in_sizes; (void)n_in; (void)out_size; (void)ws_size;
  const float* x   = (const float*)d_in[0];
  const float* Wq  = (const float*)d_in[1];
  const float* Wkv = (const float*)d_in[2];
  const float* Wo  = (const float*)d_in[3];
  float* out = (float*)d_out;

  const long XE  = (long)BATCH * SEQ * DMODEL;   // 16,777,216
  const long WQE = (long)DMODEL * DMODEL;        //  4,194,304
  const long WKE = (long)DMODEL * 2 * HDIM;      //    524,288
  const long KE  = (long)BATCH * SEQ * HDIM;     //  1,048,576

  char* ws = (char*)d_ws;
  size_t off = 0;
  auto take = [&](size_t bytes) -> char* {
    char* p = ws + off;
    off += (bytes + 255) & ~(size_t)255;
    return p;
  };
  __bf16* xb   = (__bf16*)take(XE  * 2);
  __bf16* WqT  = (__bf16*)take(WQE * 2);
  __bf16* WkvT = (__bf16*)take(WKE * 2);
  __bf16* WoT  = (__bf16*)take(WQE * 2);
  __bf16* q_b  = (__bf16*)take(XE  * 2);   // (b,h,n,hd)
  __bf16* k_b  = (__bf16*)take(KE  * 2);   // (b,t,hd)
  __bf16* vT_b = (__bf16*)take(KE  * 2);   // (b,hd,t)
  __bf16* y_b  = (__bf16*)take(XE  * 2);   // (b,n,d)

  const float scale = 0.08838834764831845f;  // 128^-0.5

  cvt_bf16_kernel<<<8192, 256, 0, stream>>>(x, xb, XE);
  cvt_transpose_kernel<<<8192, 256, 0, stream>>>(Wq,  WqT,  DMODEL, DMODEL);
  cvt_transpose_kernel<<<2048, 256, 0, stream>>>(Wkv, WkvT, DMODEL, 2 * HDIM);
  cvt_transpose_kernel<<<8192, 256, 0, stream>>>(Wo,  WoT,  DMODEL, DMODEL);

  const int M = BATCH * SEQ;  // 8192

  gemm_wmma_kernel<1><<<dim3(DMODEL / 128, M / 64), 256, 0, stream>>>(
      xb, WqT, nullptr, q_b, nullptr, M, DMODEL, DMODEL, scale);

  gemm_wmma_kernel<2><<<dim3((2 * HDIM) / 128, M / 64), 256, 0, stream>>>(
      xb, WkvT, nullptr, k_b, vT_b, M, 2 * HDIM, DMODEL, 1.0f);

  mqa_flash_kernel<<<2048, 128, 0, stream>>>(q_b, k_b, vT_b, y_b);

  gemm_wmma_kernel<0><<<dim3(DMODEL / 128, M / 64), 256, 0, stream>>>(
      y_b, WoT, out, nullptr, nullptr, M, DMODEL, DMODEL, 1.0f);
}
